// MultiQueryAttention_89275190214871
// MI455X (gfx1250) — matrix operand order repaired and checked by reading, not compiled, not run on hardware
//
#include <hip/hip_runtime.h>
#include <hip/hip_bf16.h>

typedef unsigned int   u32;
typedef unsigned short u16;
typedef __attribute__((ext_vector_type(16))) __bf16 v16bf;
typedef __attribute__((ext_vector_type(8)))  float  v8f;
typedef __attribute__((ext_vector_type(4)))  u32    u32x4;

union ABf { u32x4 q[2]; v16bf v; };

__device__ __forceinline__ u16 f2bf(float f) {
  u32 u = __builtin_bit_cast(u32, f);
  u += 0x7FFFu + ((u >> 16) & 1u);   // round-to-nearest-even
  return (u16)(u >> 16);
}

#define KDIM 1024
#define SEQ  2048
#define NROW 4096   /* B*S */
#define NHEAD 16

// ---------------- prep: f32 -> bf16 (flat) ----------------
__global__ void cvt_bf16_kernel(const float* __restrict__ src, u16* __restrict__ dst, int n) {
  int i = blockIdx.x * blockDim.x + threadIdx.x;
  if (i < n) dst[i] = f2bf(src[i]);
}

// ---------------- prep: f32 [K][N] -> bf16 [N][K] ----------------
__global__ void cvt_bf16_T_kernel(const float* __restrict__ src, u16* __restrict__ dst, int K, int N) {
  int i = blockIdx.x * blockDim.x + threadIdx.x;
  if (i < K * N) {
    int k = i / N, n = i - k * N;
    dst[(size_t)n * K + k] = f2bf(src[i]);
  }
}

// ---------------- bf16 WMMA GEMM: 32x64 tile per wave, double-buffered frags ----
// outMode: 0 = bf16 row-major [m][ldo], 1 = bf16 transposed [n][ldo], 2 = f32 row-major [m][ldo]
__global__ void gemm_bf16_wmma(const u16* __restrict__ A, const u16* __restrict__ BT,
                               const float* __restrict__ bias, void* __restrict__ out,
                               int ldo, int outMode) {
  const int lane = threadIdx.x;
  const int mrow = lane & 15;
  const int hi   = lane >> 4;
  const int n0   = blockIdx.x * 64;
  const int m0   = blockIdx.y * 32;
  const int akb  = hi * 8;    // A K-chunk base (ISA 16-bit A layout)
  const int bkb  = hi * 8;    // B K-chunk base (same layout as A)

  const u16* Ar0 = A + (size_t)(m0 + mrow) * KDIM + akb;
  const u16* Ar1 = Ar0 + (size_t)16 * KDIM;
  const u16* Bp[4];
#pragma unroll
  for (int nf = 0; nf < 4; ++nf)
    Bp[nf] = BT + (size_t)(n0 + nf * 16 + mrow) * KDIM + bkb;

  ABf a0[2], a1[2], bfr[2][4];
  auto ldfrags = [&](int ph, int k) {
    a0[ph].q[0] = *(const u32x4*)(Ar0 + k);
    a0[ph].q[1] = *(const u32x4*)(Ar0 + k + 16);
    a1[ph].q[0] = *(const u32x4*)(Ar1 + k);
    a1[ph].q[1] = *(const u32x4*)(Ar1 + k + 16);
#pragma unroll
    for (int nf = 0; nf < 4; ++nf) {
      bfr[ph][nf].q[0] = *(const u32x4*)(Bp[nf] + k);
      bfr[ph][nf].q[1] = *(const u32x4*)(Bp[nf] + k + 16);
    }
  };

  v8f c[2][4] = {};
  ldfrags(0, 0);
  for (int k = 0; k < KDIM; k += 32) {
    const int ph = (k >> 5) & 1;
    if (k + 32 < KDIM) ldfrags(ph ^ 1, k + 32);   // prefetch next slab into regs
    if (k + 128 < KDIM) {                          // warm caches further ahead
      __builtin_prefetch(Ar0 + k + 128, 0, 0);     // -> global_prefetch_b8
      __builtin_prefetch(Ar1 + k + 128, 0, 0);
    }
#pragma unroll
    for (int nf = 0; nf < 4; ++nf) {
      c[0][nf] = __builtin_amdgcn_wmma_f32_16x16x32_bf16(false, a0[ph].v, false, bfr[ph][nf].v,
                                                         (short)0, c[0][nf], false, false);
      c[1][nf] = __builtin_amdgcn_wmma_f32_16x16x32_bf16(false, a1[ph].v, false, bfr[ph][nf].v,
                                                         (short)0, c[1][nf], false, false);
    }
  }
#pragma unroll
  for (int rt = 0; rt < 2; ++rt) {
#pragma unroll
    for (int nf = 0; nf < 4; ++nf) {
#pragma unroll
      for (int r = 0; r < 8; ++r) {
        int mm = m0 + rt * 16 + r + 8 * hi;   // C/D layout: M = r + 8*(lane>=16)
        int nn = n0 + nf * 16 + mrow;         // N = lane%16
        float val = c[rt][nf][r] + bias[nn];
        if (outMode == 0)      ((u16*)out)[(size_t)mm * ldo + nn] = f2bf(val);
        else if (outMode == 1) ((u16*)out)[(size_t)nn * ldo + mm] = f2bf(val);
        else                   ((float*)out)[(size_t)mm * ldo + nn] = val;
      }
    }
  }
}

// ---------------- flash MQA attention: 1 wave = 16 query rows of one (b,h) ----
// Q:  bf16 [B*S][H*64]   Kb: bf16 [B*S][64]   VT: bf16 [64][B*S]   O: bf16 [B*S][H*64]
__global__ void mqa_attn_kernel(const u16* __restrict__ Q, const u16* __restrict__ Kb,
                                const u16* __restrict__ VT, u16* __restrict__ Obf) {
  __shared__ u16 pbuf[16 * 32];            // P tile staging (C-layout -> A-layout)
  const int lane = threadIdx.x;
  const int mrow = lane & 15;
  const int hi   = lane >> 4;
  const int s0   = blockIdx.x * 16;
  const int bh   = blockIdx.y;
  const int b    = bh >> 4;
  const int h    = bh & 15;
  const int akb  = hi * 8;
  const int bkb  = hi * 8;

  // Q fragments: 16x64 = two A frags (K=0..31, K=32..63)
  const size_t qrow = (size_t)(b * SEQ + s0 + mrow) * (NHEAD * 64) + h * 64;
  ABf qa[2];
#pragma unroll
  for (int kf = 0; kf < 2; ++kf) {
    qa[kf].q[0] = *(const u32x4*)(Q + qrow + kf * 32 + akb);
    qa[kf].q[1] = *(const u32x4*)(Q + qrow + kf * 32 + akb + 16);
  }

  // double-buffered K/V fragments for the 32-key slab
  ABf kfr[2][4], vfr[2][4];
  auto ldKV = [&](int ph, int t0) {
#pragma unroll
    for (int nt = 0; nt < 2; ++nt) {
#pragma unroll
      for (int kf = 0; kf < 2; ++kf) {
        const u16* kp = Kb + (size_t)(b * SEQ + t0 + nt * 16 + mrow) * 64 + kf * 32 + bkb;
        kfr[ph][nt * 2 + kf].q[0] = *(const u32x4*)(kp);
        kfr[ph][nt * 2 + kf].q[1] = *(const u32x4*)(kp + 16);
      }
    }
#pragma unroll
    for (int nf = 0; nf < 4; ++nf) {
      const u16* vp = VT + (size_t)(nf * 16 + mrow) * NROW + b * SEQ + t0 + bkb;
      vfr[ph][nf].q[0] = *(const u32x4*)(vp);
      vfr[ph][nf].q[1] = *(const u32x4*)(vp + 16);
    }
  };

  float m[8], l[8];
  v8f o[4] = {};
#pragma unroll
  for (int r = 0; r < 8; ++r) { m[r] = -3.0e38f; l[r] = 0.f; }

  ldKV(0, 0);
  for (int t0 = 0; t0 < SEQ; t0 += 32) {
    const int ph = (t0 >> 5) & 1;
    if (t0 + 32 < SEQ) ldKV(ph ^ 1, t0 + 32);   // prefetch next key slab into regs
    if (t0 + 128 < SEQ) {                        // warm caches further ahead
      __builtin_prefetch(Kb + (size_t)(b * SEQ + t0 + 128 + mrow) * 64, 0, 0);
      __builtin_prefetch(VT + (size_t)(mrow) * NROW + b * SEQ + t0 + 128, 0, 0);
    }

    // ---- scores: two 16x16 tiles over dk=64 ----
    v8f sc[2] = {};
#pragma unroll
    for (int nt = 0; nt < 2; ++nt) {
#pragma unroll
      for (int kf = 0; kf < 2; ++kf) {
        sc[nt] = __builtin_amdgcn_wmma_f32_16x16x32_bf16(false, qa[kf].v, false,
                                                         kfr[ph][nt * 2 + kf].v,
                                                         (short)0, sc[nt], false, false);
      }
    }
    // ---- online softmax over this 32-key slab ----
    float pn0[8], pn1[8];
#pragma unroll
    for (int r = 0; r < 8; ++r) {
      float v0 = sc[0][r] * 0.125f;        // 1/sqrt(dk)
      float v1 = sc[1][r] * 0.125f;
      float tm = fmaxf(v0, v1);
#pragma unroll
      for (int mask = 1; mask < 16; mask <<= 1) tm = fmaxf(tm, __shfl_xor(tm, mask, 32));
      float mn    = fmaxf(m[r], tm);
      float scale = __expf(m[r] - mn);
      float p0 = __expf(v0 - mn);
      float p1 = __expf(v1 - mn);
      float ts = p0 + p1;
#pragma unroll
      for (int mask = 1; mask < 16; mask <<= 1) ts += __shfl_xor(ts, mask, 32);
      l[r] = l[r] * scale + ts;
      m[r] = mn;
#pragma unroll
      for (int nf = 0; nf < 4; ++nf) o[nf][r] *= scale;
      pn0[r] = p0; pn1[r] = p1;
    }
    // ---- P: f32 C-layout -> bf16 A-layout via LDS ----
    __syncthreads();
#pragma unroll
    for (int r = 0; r < 8; ++r) {
      int row = r + 8 * hi;
      pbuf[row * 32 + mrow]      = f2bf(pn0[r]);
      pbuf[row * 32 + 16 + mrow] = f2bf(pn1[r]);
    }
    __syncthreads();
    ABf pa;
    pa.q[0] = *(const u32x4*)(pbuf + mrow * 32 + akb);
    pa.q[1] = *(const u32x4*)(pbuf + mrow * 32 + akb + 16);
    // ---- O += P(16x32) * V(32x64) ----
#pragma unroll
    for (int nf = 0; nf < 4; ++nf) {
      o[nf] = __builtin_amdgcn_wmma_f32_16x16x32_bf16(false, pa.v, false, vfr[ph][nf].v,
                                                      (short)0, o[nf], false, false);
    }
    __syncthreads();
  }
  // ---- normalize, write bf16 ----
#pragma unroll
  for (int nf = 0; nf < 4; ++nf) {
#pragma unroll
    for (int r = 0; r < 8; ++r) {
      int mm = s0 + r + 8 * hi;
      float val = o[nf][r] / l[r];
      Obf[(size_t)(b * SEQ + mm) * (NHEAD * 64) + h * 64 + nf * 16 + mrow] = f2bf(val);
    }
  }
}

extern "C" void kernel_launch(void* const* d_in, const int* in_sizes, int n_in,
                              void* d_out, int out_size, void* d_ws, size_t ws_size,
                              hipStream_t stream) {
  const float* x  = (const float*)d_in[0];
  const float* Wq = (const float*)d_in[1];
  const float* bq = (const float*)d_in[2];
  const float* Wk = (const float*)d_in[3];
  const float* bk = (const float*)d_in[4];
  const float* Wv = (const float*)d_in[5];
  const float* bv = (const float*)d_in[6];
  const float* Wo = (const float*)d_in[7];
  const float* bo = (const float*)d_in[8];

  char* ws = (char*)d_ws;
  size_t off = 0;
  auto carve = [&](size_t bytes) { char* p = ws + off; off += (bytes + 255) & ~(size_t)255; return p; };
  u16* xb  = (u16*)carve((size_t)NROW * KDIM * 2);      // x in bf16
  u16* wqT = (u16*)carve((size_t)1024 * 1024 * 2);      // Wq^T bf16 [1024][1024]
  u16* wkT = (u16*)carve((size_t)64 * 1024 * 2);        // Wk^T bf16 [64][1024]
  u16* wvT = (u16*)carve((size_t)64 * 1024 * 2);        // Wv^T bf16 [64][1024]
  u16* woT = (u16*)carve((size_t)1024 * 1024 * 2);      // Wo^T bf16 [1024][1024]
  u16* qb  = (u16*)carve((size_t)NROW * 1024 * 2);      // q bf16 [B*S][H*64]
  u16* kb  = (u16*)carve((size_t)NROW * 64 * 2);        // k bf16 [B*S][64]
  u16* vT  = (u16*)carve((size_t)64 * NROW * 2);        // v^T bf16 [64][B*S]
  u16* ab  = (u16*)carve((size_t)NROW * 1024 * 2);      // attention out bf16

  const int nx = NROW * KDIM;
  cvt_bf16_kernel<<<(nx + 255) / 256, 256, 0, stream>>>(x, xb, nx);
  cvt_bf16_T_kernel<<<(1024 * 1024 + 255) / 256, 256, 0, stream>>>(Wq, wqT, 1024, 1024);
  cvt_bf16_T_kernel<<<(1024 * 64 + 255) / 256, 256, 0, stream>>>(Wk, wkT, 1024, 64);
  cvt_bf16_T_kernel<<<(1024 * 64 + 255) / 256, 256, 0, stream>>>(Wv, wvT, 1024, 64);
  cvt_bf16_T_kernel<<<(1024 * 1024 + 255) / 256, 256, 0, stream>>>(Wo, woT, 1024, 1024);

  // q = x*Wq + bq -> bf16 [4096][1024]
  gemm_bf16_wmma<<<dim3(16, 128), 32, 0, stream>>>(xb, wqT, bq, qb, 1024, 0);
  // k = x*Wk + bk -> bf16 [4096][64]
  gemm_bf16_wmma<<<dim3(1, 128), 32, 0, stream>>>(xb, wkT, bk, kb, 64, 0);
  // v = x*Wv + bv -> bf16 transposed [64][4096]
  gemm_bf16_wmma<<<dim3(1, 128), 32, 0, stream>>>(xb, wvT, bv, vT, NROW, 1);
  // flash attention per (16 q-rows, b, h)
  mqa_attn_kernel<<<dim3(SEQ / 16, 2 * NHEAD), 32, 0, stream>>>(qb, kb, vT, ab);
  // out = attn*Wo + bo -> f32 d_out
  gemm_bf16_wmma<<<dim3(16, 128), 32, 0, stream>>>(ab, woT, bo, d_out, 1024, 2);
}
